// GNN_42047729828248
// MI455X (gfx1250) — matrix operand order repaired and checked by reading, not compiled, not run on hardware
//
#include <hip/hip_runtime.h>
#include <math.h>

#define NNODES 100000
#define NEDGES 3200000
#define INDIM  256
#define HIDDIM 256
#define OUTDIM 64
#define EXP_NEG1 0.36787944117144233f
#define SELU_SCALE 1.0507009873554805f
#define SELU_ALPHA 1.6732632423543772f

typedef __attribute__((ext_vector_type(16))) _Float16 v16h;
typedef __attribute__((ext_vector_type(8)))  float    v8f;

__device__ __forceinline__ void atomAddF(float* p, float v) {
    // hardware global_atomic_add_f32 (no CAS loop)
    unsafeAtomicAdd(p, v);
}

// ---------------- degree / gcn-norm ----------------
__global__ void k_degree(const int* __restrict__ dst, float* __restrict__ deg, int E) {
    int e = blockIdx.x * blockDim.x + threadIdx.x;
    if (e < E) atomAddF(&deg[dst[e]], 1.0f);
}

__global__ void k_dinv(const float* __restrict__ deg, float* __restrict__ dinv,
                       float* __restrict__ dinv2, int N) {
    int i = blockIdx.x * blockDim.x + threadIdx.x;
    if (i < N) {
        float d = deg[i];
        dinv[i]  = d > 0.f ? rsqrtf(d) : 0.f;   // conv1: no self loops
        dinv2[i] = rsqrtf(d + 1.f);             // conv2: self loops added
    }
}

// precompute symmetric gcn edge weights once; reused across all 3 scatter passes
__global__ void k_edgew(const int* __restrict__ src, const int* __restrict__ dst,
                        const float* __restrict__ dinv, const float* __restrict__ dinv2,
                        float* __restrict__ w1, float* __restrict__ w2, int E) {
    int e = blockIdx.x * blockDim.x + threadIdx.x;
    if (e < E) {
        int s = src[e], d = dst[e];
        w1[e] = dinv[d]  * dinv[s];
        w2[e] = dinv2[d] * dinv2[s];
    }
}

// ---------------- propagation: one wave per edge, 256 feats ----------------
__global__ void k_prop(const int* __restrict__ src, const int* __restrict__ dst,
                       const float* __restrict__ wE, const float* __restrict__ xin,
                       float* __restrict__ xout, float coeff, int E) {
    int wave = (int)((blockIdx.x * (unsigned)blockDim.x + threadIdx.x) >> 5);
    int lane = threadIdx.x & 31;
    if (wave >= E) return;
    float w = wE[wave] * coeff;
    if (w == 0.f) return;
    int s = src[wave], d = dst[wave];
    const float4* xs = (const float4*)(xin + (size_t)s * INDIM);
    float*        xd = xout + (size_t)d * INDIM;
    float4 a = xs[lane];        // floats [lane*4 .. lane*4+3]
    float4 b = xs[lane + 32];   // floats [128 + lane*4 ..]
    int base = lane * 4;
    atomAddF(xd + base + 0, w * a.x);
    atomAddF(xd + base + 1, w * a.y);
    atomAddF(xd + base + 2, w * a.z);
    atomAddF(xd + base + 3, w * a.w);
    atomAddF(xd + 128 + base + 0, w * b.x);
    atomAddF(xd + 128 + base + 1, w * b.y);
    atomAddF(xd + 128 + base + 2, w * b.z);
    atomAddF(xd + 128 + base + 3, w * b.w);
}

// self-loop term of conv2 (runs after edge scatter; stream-serialized, no atomics)
__global__ void k_selfloop(const float* __restrict__ H, const float* __restrict__ dinv2,
                           float* __restrict__ AGG, int total) {
    int idx = blockIdx.x * blockDim.x + threadIdx.x;
    if (idx < total) {
        int node = idx >> 8;
        float w = dinv2[node];
        AGG[idx] += (w * w) * H[idx];
    }
}

// ---------------- pack weights into WMMA B-fragment-major f16 layout ----------------
// Fragment (kt, nt): 32 lanes x 16 halves contiguous; lane r=lane&15 -> col, sel=lane>>4
// element e -> K = kt*32 + (e>>3)*16 + sel*8 + (e&7); N = nt*16 + r  (matches V_WMMA 32x16 f16 B layout)
__global__ void k_packB(const float* __restrict__ W, _Float16* __restrict__ Bp,
                        int K, int N) {
    int t = blockIdx.x * blockDim.x + threadIdx.x;
    int nTiles = N >> 4;
    int total = (K >> 5) * nTiles * 512;
    if (t >= total) return;
    int frag   = t >> 9;
    int within = t & 511;
    int lane = within >> 4;
    int e    = within & 15;
    int kt = frag / nTiles, nt = frag - kt * nTiles;
    int r = lane & 15, sel = lane >> 4;
    int k = kt * 32 + ((e >> 3) << 4) + (sel << 3) + (e & 7);
    int n = nt * 16 + r;
    Bp[t] = (_Float16)W[k * N + n];
}

// ---------------- GEMM1: H = selu(X @ W1 + b1), wave per 16x128 block ------------
// 8 accumulators (64 VGPRs): no spills, A panel read twice total, 8 WMMAs per A frag.
__global__ void k_gemm1_selu(const float* __restrict__ X, const _Float16* __restrict__ Bp,
                             const float* __restrict__ bias, float* __restrict__ H, int M) {
    int wave = (int)((blockIdx.x * (unsigned)blockDim.x + threadIdx.x) >> 5);
    int lane = threadIdx.x & 31;
    if (wave >= (M >> 4) * 2) return;              // wave-uniform: EXEC all-ones at WMMA
    int m0    = (wave >> 1) * 16;
    int ntBase = (wave & 1) * 8;                   // n-half: tiles [ntBase, ntBase+8)
    int r = lane & 15, sel = lane >> 4;
    const float* arow = X + (size_t)(m0 + r) * INDIM;
    v8f acc[8];
#pragma unroll
    for (int i = 0; i < 8; ++i) acc[i] = v8f{};
#pragma unroll
    for (int kt = 0; kt < 8; ++kt) {
        int k0 = kt * 32;
        v16h a;
#pragma unroll
        for (int e = 0; e < 16; ++e) {
            int ka = k0 + ((e >> 3) << 4) + (sel << 3) + (e & 7);   // A 16x32 f16 layout
            a[e] = (_Float16)arow[ka];
        }
        const _Float16* fr = Bp + ((size_t)(kt * 16 + ntBase)) * 512 + lane * 16;
#pragma unroll
        for (int i = 0; i < 8; ++i) {
            v16h b = *(const v16h*)(fr + i * 512);                  // 2x global_load_b128
            acc[i] = __builtin_amdgcn_wmma_f32_16x16x32_f16(false, a, false, b,
                                                            (short)0, acc[i], false, false);
        }
    }
    int rowbase = m0 + (sel << 3);
#pragma unroll
    for (int i = 0; i < 8; ++i) {
        int n = (ntBase + i) * 16 + r;
        float bn = bias[n];
#pragma unroll
        for (int v = 0; v < 8; ++v) {
            float val = acc[i][v] + bn;
            val = val > 0.f ? SELU_SCALE * val
                            : SELU_SCALE * SELU_ALPHA * (__expf(val) - 1.f);
            H[(size_t)(rowbase + v) * HIDDIM + n] = val;
        }
    }
}

// ---------------- GEMM2: OUT = X @ W2 + b2, wave per 16x64 row-block ----------------
__global__ void k_gemm2(const float* __restrict__ X, const _Float16* __restrict__ Bp,
                        const float* __restrict__ bias, float* __restrict__ OUT, int M) {
    int wave = (int)((blockIdx.x * (unsigned)blockDim.x + threadIdx.x) >> 5);
    int lane = threadIdx.x & 31;
    if (wave >= (M >> 4)) return;                  // wave-uniform guard
    int m0 = wave * 16;
    int r = lane & 15, sel = lane >> 4;
    const float* arow = X + (size_t)(m0 + r) * HIDDIM;
    v8f acc[4] = {v8f{}, v8f{}, v8f{}, v8f{}};
#pragma unroll
    for (int kt = 0; kt < 8; ++kt) {
        int k0 = kt * 32;
        v16h a;
#pragma unroll
        for (int e = 0; e < 16; ++e) {
            int ka = k0 + ((e >> 3) << 4) + (sel << 3) + (e & 7);
            a[e] = (_Float16)arow[ka];
        }
        const _Float16* fr = Bp + ((size_t)kt * 4) * 512 + lane * 16;
#pragma unroll
        for (int nt = 0; nt < 4; ++nt) {           // A-fragment reused 4x
            v16h b = *(const v16h*)(fr + nt * 512);
            acc[nt] = __builtin_amdgcn_wmma_f32_16x16x32_f16(false, a, false, b,
                                                             (short)0, acc[nt], false, false);
        }
    }
#pragma unroll
    for (int nt = 0; nt < 4; ++nt) {
        int n = nt * 16 + r;
        float bn = bias[n];
#pragma unroll
        for (int v = 0; v < 8; ++v) {
            int row = m0 + (sel << 3) + v;
            OUT[(size_t)row * OUTDIM + n] = acc[nt][v] + bn;
        }
    }
}

// ---------------- in-place log_softmax, one wave per row of 64 ----------------
__global__ void k_logsoftmax(float* __restrict__ OUT, int M) {
    int wave = (int)((blockIdx.x * (unsigned)blockDim.x + threadIdx.x) >> 5);
    int lane = threadIdx.x & 31;
    if (wave >= M) return;
    float* row = OUT + (size_t)wave * OUTDIM;
    float v0 = row[lane], v1 = row[lane + 32];
    float m = fmaxf(v0, v1);
#pragma unroll
    for (int off = 16; off > 0; off >>= 1) m = fmaxf(m, __shfl_xor(m, off, 32));
    float s = __expf(v0 - m) + __expf(v1 - m);
#pragma unroll
    for (int off = 16; off > 0; off >>= 1) s += __shfl_xor(s, off, 32);
    float lz = m + __logf(s);
    row[lane]      = v0 - lz;
    row[lane + 32] = v1 - lz;
}

static inline size_t alignUp(size_t x) { return (x + 255) & ~(size_t)255; }

extern "C" void kernel_launch(void* const* d_in, const int* in_sizes, int n_in,
                              void* d_out, int out_size, void* d_ws, size_t ws_size,
                              hipStream_t stream) {
    const float* x    = (const float*)d_in[0];
    const int*   esrc = (const int*)d_in[1];
    const int*   edst = (const int*)d_in[2];
    const float* W1   = (const float*)d_in[3];
    const float* b1   = (const float*)d_in[4];
    const float* W2   = (const float*)d_in[5];
    const float* b2   = (const float*)d_in[6];
    float* out = (float*)d_out;

    const size_t FEAT_BYTES = (size_t)NNODES * INDIM * sizeof(float);
    char* p = (char*)d_ws;
    float* bufA = (float*)p; p += alignUp(FEAT_BYTES);   // x1, then H
    float* bufB = (float*)p; p += alignUp(FEAT_BYTES);   // x2, then H_agg
    float* deg  = (float*)p; p += alignUp(NNODES * sizeof(float));
    float* dinv = (float*)p; p += alignUp(NNODES * sizeof(float));
    float* dnv2 = (float*)p; p += alignUp(NNODES * sizeof(float));
    float* w1E  = (float*)p; p += alignUp(NEDGES * sizeof(float));
    float* w2E  = (float*)p; p += alignUp(NEDGES * sizeof(float));
    _Float16* B1p = (_Float16*)p; p += alignUp(INDIM * HIDDIM * sizeof(_Float16));
    _Float16* B2p = (_Float16*)p; p += alignUp(HIDDIM * OUTDIM * sizeof(_Float16));

    hipMemsetAsync(deg,  0, NNODES * sizeof(float), stream);
    hipMemsetAsync(bufA, 0, FEAT_BYTES, stream);
    hipMemsetAsync(bufB, 0, FEAT_BYTES, stream);

    k_packB<<<(INDIM * HIDDIM + 255) / 256, 256, 0, stream>>>(W1, B1p, INDIM, HIDDIM);
    k_packB<<<(HIDDIM * OUTDIM + 255) / 256, 256, 0, stream>>>(W2, B2p, HIDDIM, OUTDIM);
    k_degree<<<(NEDGES + 255) / 256, 256, 0, stream>>>(edst, deg, NEDGES);
    k_dinv<<<(NNODES + 255) / 256, 256, 0, stream>>>(deg, dinv, dnv2, NNODES);
    k_edgew<<<(NEDGES + 255) / 256, 256, 0, stream>>>(esrc, edst, dinv, dnv2, w1E, w2E, NEDGES);

    const int propBlocks = (NEDGES + 7) / 8;  // 8 waves (one edge each) per 256-thread block
    // conv1 hop 1: x1 = S x
    k_prop<<<propBlocks, 256, 0, stream>>>(esrc, edst, w1E, x, bufA, 1.0f, NEDGES);
    // conv1 hop 2: x2 = e^-1 * S x1
    k_prop<<<propBlocks, 256, 0, stream>>>(esrc, edst, w1E, bufA, bufB, EXP_NEG1, NEDGES);

    // H = selu(x2 @ W1 + b1)  (bufB -> bufA), one wave per 16x128 half-panel
    const int g1waves = (NNODES / 16) * 2;                      // 12500
    k_gemm1_selu<<<(g1waves * 32 + 255) / 256, 256, 0, stream>>>(bufB, B1p, b1, bufA, NNODES);

    // conv2 aggregation with self loops (bufA -> bufB)
    hipMemsetAsync(bufB, 0, FEAT_BYTES, stream);
    k_prop<<<propBlocks, 256, 0, stream>>>(esrc, edst, w2E, bufA, bufB, 1.0f, NEDGES);
    k_selfloop<<<(NNODES * INDIM + 255) / 256, 256, 0, stream>>>(bufA, dnv2, bufB, NNODES * INDIM);

    // logits = H_agg @ W2 + b2 -> d_out, then in-place log_softmax
    const int g2waves = NNODES / 16;                            // 6250
    k_gemm2<<<(g2waves * 32 + 255) / 256, 256, 0, stream>>>(bufB, B2p, b2, out, NNODES);
    k_logsoftmax<<<(NNODES * 32 + 255) / 256, 256, 0, stream>>>(out, NNODES);
}
